// TransformerBlock_68212670595278
// MI455X (gfx1250) — matrix operand order repaired and checked by reading, not compiled, not run on hardware
//
#include <hip/hip_runtime.h>
#include <cstdint>
#include <cstddef>

// ---------------------------------------------------------------------------
// Types for CDNA5 WMMA (wave32): D(16x16 f32) = A(16x32 bf16) x B(32x16 bf16) + C
// ---------------------------------------------------------------------------
typedef __attribute__((ext_vector_type(16))) __bf16       bf16x16;
typedef __attribute__((ext_vector_type(8)))  float        f32x8;
typedef __attribute__((ext_vector_type(4)))  int          i32x4;
typedef __attribute__((ext_vector_type(8)))  int          i32x8;
typedef __attribute__((ext_vector_type(4)))  unsigned int u32x4;

union FragU {
    bf16x16 v;
    uint4   u[2];
    __bf16  e[16];
};

__device__ __forceinline__ f32x8 wmma_bf16(bf16x16 a, bf16x16 b, f32x8 c) {
    return __builtin_amdgcn_wmma_f32_16x16x32_bf16(
        /*neg_a=*/false, a, /*neg_b=*/false, b,
        /*c_mod=*/(short)0, c, /*reuse_a=*/false, /*reuse_b=*/false);
}

// ---------------------------------------------------------------------------
// CDNA5 data movers: TDM (TENSOR_LOAD_TO_LDS, TENSORcnt) preferred,
// async global->LDS (GLOBAL_LOAD_ASYNC_TO_LDS_B128, ASYNCcnt) as fallback,
// plain load+ds_store as last resort.
// ---------------------------------------------------------------------------
#if __has_builtin(__builtin_amdgcn_tensor_load_to_lds) && \
    __has_builtin(__builtin_amdgcn_s_wait_tensorcnt)
#define USE_TDM 1
#else
#define USE_TDM 0
#endif

#if __has_builtin(__builtin_amdgcn_global_load_async_to_lds_b128) && \
    __has_builtin(__builtin_amdgcn_s_wait_asynccnt)
#define USE_ASYNC_LDS 1
#else
#define USE_ASYNC_LDS 0
#endif

#if USE_ASYNC_LDS
typedef __attribute__((address_space(1))) i32x4* gv4i_p;
typedef __attribute__((address_space(3))) i32x4* lv4i_p;
#endif

__device__ __forceinline__ void copy16_g2l(__bf16* ldst, const __bf16* gsrc) {
#if USE_ASYNC_LDS
    __builtin_amdgcn_global_load_async_to_lds_b128(
        (gv4i_p)(uintptr_t)gsrc, (lv4i_p)(uint32_t)(uintptr_t)ldst, 0, 0);
#else
    *reinterpret_cast<uint4*>(ldst) = *reinterpret_cast<const uint4*>(gsrc);
#endif
}

template <int N>
__device__ __forceinline__ void async_wait_le() {
#if USE_ASYNC_LDS
    __builtin_amdgcn_s_wait_asynccnt(N);
#endif
}

// Issue one 2-D bf16 tile DMA: tile_rows x tile_cols from a row-major tensor
// (row stride = row_stride_elems), into LDS at lds_addr with row padding
// controlled by pad_interval_enc (pad after 2<<enc dwords) and pad_amount_enc
// (enc+1 dwords of pad).  D# layout per CDNA5 ISA 8.3/8.4.
// Toolchain builtin is the 6-arg form: (g0, g1, g2, g3, g_ext, cpol).
__device__ __forceinline__ void tdm_load_2d_bf16(const void* gsrc, unsigned lds_addr,
                                                 unsigned tile_cols, unsigned tile_rows,
                                                 unsigned long long row_stride_elems,
                                                 unsigned pad_interval_enc,
                                                 unsigned pad_amount_enc) {
#if USE_TDM
    const unsigned long long ga = (unsigned long long)(uintptr_t)gsrc;
    u32x4 g0;
    g0[0] = 1u;                                            // count=1, user mode
    g0[1] = lds_addr;                                      // LDS byte address
    g0[2] = (unsigned)ga;                                  // global addr [31:0]
    g0[3] = (unsigned)((ga >> 32) & 0x1FFFFFFu) | (2u << 30); // addr[56:32] | type=2
    const unsigned w0 = (1u << 16)                         // data_size = 2 bytes
                      | (1u << 20)                         // pad_enable
                      | ((pad_interval_enc & 7u) << 22)
                      | ((pad_amount_enc & 127u) << 25);
    const unsigned tdim0 = (unsigned)row_stride_elems;     // tensor dim0 (>= tile_cols)
    const unsigned tdim1 = 0x7FFFFFFFu;                    // effectively unbounded rows
    i32x8 g1;
    g1[0] = (int)w0;
    g1[1] = (int)((tdim0 & 0xFFFFu) << 16);                // abar=0 | tdim0[15:0]
    g1[2] = (int)(((tdim0 >> 16) & 0xFFFFu) | ((tdim1 & 0xFFFFu) << 16));
    g1[3] = (int)(((tdim1 >> 16) & 0xFFFFu) | ((tile_cols & 0xFFFFu) << 16));
    g1[4] = (int)(tile_rows & 0xFFFFu);                    // tile_dim1 | tile_dim2=0
    g1[5] = (int)(unsigned)row_stride_elems;               // dim0 stride [31:0]
    g1[6] = (int)(unsigned)((row_stride_elems >> 32) & 0xFFFFu);
    g1[7] = 0;
    const i32x4 z4 = {0, 0, 0, 0};
    const i32x8 z8 = {0, 0, 0, 0, 0, 0, 0, 0};
    __builtin_amdgcn_tensor_load_to_lds(g0, g1, z4, z4, z8, 0);
#else
    (void)gsrc; (void)lds_addr; (void)tile_cols; (void)tile_rows;
    (void)row_stride_elems; (void)pad_interval_enc; (void)pad_amount_enc;
#endif
}

template <int N>
__device__ __forceinline__ void tensor_wait_le() {
#if USE_TDM
    __builtin_amdgcn_s_wait_tensorcnt(N);
#endif
}

// ---------------------------------------------------------------------------
// Fragment loaders
// ---------------------------------------------------------------------------
// A-matrix fragment (16x32, MxK) from row-major bf16, leading dim ld.
__device__ __forceinline__ bf16x16 load_a_bf16(const __bf16* base, size_t ld,
                                               size_t row, int k0, int lane) {
    const int lo = (lane & 16) ? 16 : 0;
    const __bf16* p = base + row * ld + (size_t)(k0 + lo);
    FragU r;
    r.u[0] = *reinterpret_cast<const uint4*>(p);
    r.u[1] = *reinterpret_cast<const uint4*>(p + 8);
    return r.v;
}

// A fragment from an LDS row (contiguous 32 K-values at row_ptr).
__device__ __forceinline__ bf16x16 load_a_lds(const __bf16* row_ptr, int lane) {
    const int lo = (lane & 16) ? 16 : 0;
    FragU r;
    r.u[0] = *reinterpret_cast<const uint4*>(row_ptr + lo);
    r.u[1] = *reinterpret_cast<const uint4*>(row_ptr + 8 + lo);
    return r.v;
}

// B-matrix fragment (32x16, KxN) from N-major bf16 (row = n, contiguous k).
__device__ __forceinline__ bf16x16 load_b_bf16(const __bf16* base, size_t ld,
                                               size_t col, int k0, int lane) {
    const int lo = (lane & 16) ? 16 : 0;
    const __bf16* p = base + col * ld + (size_t)(k0 + lo);
    FragU r;
    r.u[0] = *reinterpret_cast<const uint4*>(p);
    r.u[1] = *reinterpret_cast<const uint4*>(p + 8);
    return r.v;
}

// B fragment from one N-major row already resident in LDS (contiguous k).
__device__ __forceinline__ bf16x16 load_b_row(const __bf16* row_ptr, int k0, int lane) {
    const int lo = (lane & 16) ? 16 : 0;
    const __bf16* p = row_ptr + k0 + lo;
    FragU r;
    r.u[0] = *reinterpret_cast<const uint4*>(p);
    r.u[1] = *reinterpret_cast<const uint4*>(p + 8);
    return r.v;
}

// B fragment from fp32 weights W[n,k] (row-major, ld = K); cvt to bf16 in regs.
__device__ __forceinline__ bf16x16 load_b_f32(const float* W, size_t ld,
                                              size_t col, int k0, int lane) {
    const int lo = (lane & 16) ? 16 : 0;
    const float* p = W + col * ld + (size_t)(k0 + lo);
    float4 f0 = *reinterpret_cast<const float4*>(p);
    float4 f1 = *reinterpret_cast<const float4*>(p + 4);
    float4 f2 = *reinterpret_cast<const float4*>(p + 8);
    float4 f3 = *reinterpret_cast<const float4*>(p + 12);
    FragU r;
    r.e[0]  = (__bf16)f0.x; r.e[1]  = (__bf16)f0.y; r.e[2]  = (__bf16)f0.z; r.e[3]  = (__bf16)f0.w;
    r.e[4]  = (__bf16)f1.x; r.e[5]  = (__bf16)f1.y; r.e[6]  = (__bf16)f1.z; r.e[7]  = (__bf16)f1.w;
    r.e[8]  = (__bf16)f2.x; r.e[9]  = (__bf16)f2.y; r.e[10] = (__bf16)f2.z; r.e[11] = (__bf16)f2.w;
    r.e[12] = (__bf16)f3.x; r.e[13] = (__bf16)f3.y; r.e[14] = (__bf16)f3.z; r.e[15] = (__bf16)f3.w;
    return r.v;
}

// B fragment from a column-strided bf16 source (global or LDS), pitch in elems.
__device__ __forceinline__ bf16x16 load_b_strided(const __bf16* base, size_t ld, int lane) {
    const int lo = (lane & 16) ? 16 : 0;
    FragU r;
#pragma unroll
    for (int j = 0; j < 16; ++j) r.e[j] = base[(size_t)(lo + j) * ld];
    return r.v;
}

// ---------------------------------------------------------------------------
// LayerNorm: one row (length Dm) per 256-thread block; writes bf16.
// ---------------------------------------------------------------------------
__global__ __launch_bounds__(256) void ln_kernel(const float* __restrict__ x,
                                                 const float* __restrict__ g,
                                                 const float* __restrict__ bt,
                                                 __bf16* __restrict__ out, int Dm) {
    __shared__ float red[256];
    const int row = blockIdx.x, tid = threadIdx.x;
    const float* xr = x + (size_t)row * Dm;

    float s = 0.f;
    for (int i = tid; i < Dm; i += 256) s += xr[i];
    red[tid] = s; __syncthreads();
    for (int off = 128; off > 0; off >>= 1) {
        if (tid < off) red[tid] += red[tid + off];
        __syncthreads();
    }
    const float mu = red[0] / (float)Dm;
    __syncthreads();

    float v2 = 0.f;
    for (int i = tid; i < Dm; i += 256) { float d = xr[i] - mu; v2 += d * d; }
    red[tid] = v2; __syncthreads();
    for (int off = 128; off > 0; off >>= 1) {
        if (tid < off) red[tid] += red[tid + off];
        __syncthreads();
    }
    const float rstd = rsqrtf(red[0] / (float)Dm + 1e-5f);

    __bf16* orow = out + (size_t)row * Dm;
    for (int i = tid; i < Dm; i += 256)
        orow[i] = (__bf16)((xr[i] - mu) * rstd * g[i] + bt[i]);
}

// ---------------------------------------------------------------------------
// NT GEMM: Y[m,n] = sum_k A[m,k] * W[n,k].  A bf16 [M,K], W fp32 [N,K].
// Block = 4 waves, block tile 128x128, wave tile 64x64.
// A-tile (128x32 bf16) staged into double-buffered LDS via TDM (or async
// copies); W converted fp32->bf16 in registers (fp32 weights live in L2).
// MODE 0: Y->bf16.  1: relu(Y)->bf16.  2: outF = resid + Y.  3: outF += Y.
// ---------------------------------------------------------------------------
template <int MODE>
__global__ __launch_bounds__(128) void gemm_nt(const __bf16* __restrict__ A,
                                               const float* __restrict__ W,
                                               int M, int N, int K,
                                               const float* __restrict__ resid,
                                               float* __restrict__ outF,
                                               __bf16* __restrict__ outB) {
    const int lane = threadIdx.x;
    const int wid  = threadIdx.y;
    const int tid  = wid * 32 + lane;
    const int bm = blockIdx.y * 128;
    const int bn = blockIdx.x * 128;
    const int wm = (wid >> 1) * 64;
    const int wn = (wid & 1) * 64;
    (void)tid;

    __shared__ __bf16 atile[2][128][40];     // 64B rows + 16B TDM pad = 80B pitch

    f32x8 acc[4][4];
#pragma unroll
    for (int i = 0; i < 4; ++i)
#pragma unroll
        for (int j = 0; j < 4; ++j)
#pragma unroll
            for (int e = 0; e < 8; ++e) acc[i][j][e] = 0.f;

#if USE_TDM
    // One descriptor per 128x32 tile; wave 0 drives the DMA engine.
    auto issue_tile = [&](int k0, int buf) {
        if (wid == 0)
            tdm_load_2d_bf16(A + (size_t)bm * (size_t)K + k0,
                             (unsigned)(uintptr_t)&atile[buf][0][0],
                             /*cols=*/32, /*rows=*/128,
                             (unsigned long long)K,
                             /*pad after 16 dw=*/3, /*pad 4 dw=*/3);
    };
#else
    auto issue_tile = [&](int k0, int buf) {
        const __bf16* g = A + (size_t)(bm + tid) * (size_t)K + k0;
        __bf16* l = &atile[buf][tid][0];
#pragma unroll
        for (int c = 0; c < 4; ++c) copy16_g2l(l + 8 * c, g + 8 * c);
    };
#endif

    const int nk = K / 32;
    issue_tile(0, 0);

    size_t colB[4];
#pragma unroll
    for (int j = 0; j < 4; ++j) colB[j] = (size_t)(bn + wn + 16 * j + (lane & 15));

    for (int kt = 0; kt < nk; ++kt) {
        const int cur = kt & 1;
        if (kt + 1 < nk) {
            issue_tile(32 * (kt + 1), cur ^ 1);
#if USE_TDM
            if (wid == 0) tensor_wait_le<1>();  // tile kt done; kt+1 in flight
#else
            async_wait_le<4>();
#endif
        } else {
#if USE_TDM
            if (wid == 0) tensor_wait_le<0>();
#else
            async_wait_le<0>();
#endif
        }
        __syncthreads();

        bf16x16 a[4], b[4];
#pragma unroll
        for (int i = 0; i < 4; ++i)
            a[i] = load_a_lds(&atile[cur][wm + 16 * i + (lane & 15)][0], lane);
#pragma unroll
        for (int j = 0; j < 4; ++j) {
            b[j] = load_b_f32(W, (size_t)K, colB[j], 32 * kt, lane);
            if (kt + 1 < nk)
                __builtin_prefetch(W + colB[j] * (size_t)K + 32 * (kt + 1), 0, 1);
        }
#pragma unroll
        for (int i = 0; i < 4; ++i)
#pragma unroll
            for (int j = 0; j < 4; ++j)
                acc[i][j] = wmma_bf16(a[i], b[j], acc[i][j]);

        __syncthreads();
    }

    const int mOff = 8 * (lane >> 4);
    const int nOff = (lane & 15);
#pragma unroll
    for (int i = 0; i < 4; ++i)
#pragma unroll
        for (int j = 0; j < 4; ++j)
#pragma unroll
            for (int r = 0; r < 8; ++r) {
                const size_t m = (size_t)(bm + wm + 16 * i + r + mOff);
                const size_t n = (size_t)(bn + wn + 16 * j + nOff);
                const size_t idx = m * (size_t)N + n;
                const float v = acc[i][j][r];
                if (MODE == 0) outB[idx] = (__bf16)v;
                if (MODE == 1) outB[idx] = (__bf16)(v > 0.f ? v : 0.f);
                if (MODE == 2) outF[idx] = resid[idx] + v;
                if (MODE == 3) outF[idx] += v;
            }
}

// ---------------------------------------------------------------------------
// Flash attention: one (batch, head, 64 q-rows) per 128-thread block.
// Each wave owns 16 q rows; head_dim = 128; key blocks of 32.
// K/V tiles (32x128 bf16, 272B padded pitch) staged into double-buffered LDS
// via TDM; scores & P@V via WMMA; online softmax via wave32 shfl_xor.
// ---------------------------------------------------------------------------
__global__ __launch_bounds__(128) void attn_kernel(const __bf16* __restrict__ Qm,
                                                   const __bf16* __restrict__ Km,
                                                   const __bf16* __restrict__ Vm,
                                                   __bf16* __restrict__ Om,
                                                   int S, int Dm) {
    const int hd   = 128;
    const int lane = threadIdx.x;
    const int wid  = threadIdx.y;
    const int qb   = blockIdx.x * 64 + wid * 16;
    const int h    = blockIdx.y;
    const int b    = blockIdx.z;

    __shared__ __bf16 plds[4][16][40];
#if USE_TDM
    __shared__ __bf16 ktile[2][32][136];    // 256B rows + 16B TDM pad = 272B pitch
    __shared__ __bf16 vtile[2][32][136];
    auto issue_kv = [&](int jb, int buf) {
        if (wid == 0) {
            const __bf16* kg = Km + ((size_t)b * S + jb) * Dm + h * hd;
            const __bf16* vg = Vm + ((size_t)b * S + jb) * Dm + h * hd;
            tdm_load_2d_bf16(kg, (unsigned)(uintptr_t)&ktile[buf][0][0],
                             /*cols=*/128, /*rows=*/32, (unsigned long long)Dm,
                             /*pad after 64 dw=*/5, /*pad 4 dw=*/3);
            tdm_load_2d_bf16(vg, (unsigned)(uintptr_t)&vtile[buf][0][0],
                             128, 32, (unsigned long long)Dm, 5, 3);
        }
    };
#endif

    const __bf16* Qh = Qm + h * hd;
#if !USE_TDM
    const __bf16* Kh = Km + h * hd;
#endif

    bf16x16 qf[4];
    const size_t qrow = (size_t)b * S + qb + (lane & 15);
#pragma unroll
    for (int kk = 0; kk < 4; ++kk)
        qf[kk] = load_a_bf16(Qh, (size_t)Dm, qrow, 32 * kk, lane);

    f32x8 accO[8];
    float mrow[8], lrow[8];
#pragma unroll
    for (int t = 0; t < 8; ++t)
#pragma unroll
        for (int e = 0; e < 8; ++e) accO[t][e] = 0.f;
#pragma unroll
    for (int r = 0; r < 8; ++r) { mrow[r] = -1e30f; lrow[r] = 0.f; }

    const float sc = 0.08838834764831845f;      // 1/sqrt(128)
    const int nj = S / 32;
#if USE_TDM
    issue_kv(0, 0);
#endif

    for (int it = 0; it < nj; ++it) {
        const int jb = 32 * it;
        const int cur = it & 1;
        (void)cur; (void)jb;
#if USE_TDM
        if (it + 1 < nj) {
            issue_kv(32 * (it + 1), cur ^ 1);
            if (wid == 0) tensor_wait_le<2>();  // tile it's two DMAs done
        } else {
            if (wid == 0) tensor_wait_le<0>();
        }
        __syncthreads();
#endif

        // ---- scores S = Q K^T * sc (two 16-col tiles) ----
        f32x8 s0, s1;
#pragma unroll
        for (int e = 0; e < 8; ++e) { s0[e] = 0.f; s1[e] = 0.f; }
#if USE_TDM
#pragma unroll
        for (int kk = 0; kk < 4; ++kk) {
            bf16x16 kf0 = load_b_row(&ktile[cur][lane & 15][0],        32 * kk, lane);
            bf16x16 kf1 = load_b_row(&ktile[cur][16 + (lane & 15)][0], 32 * kk, lane);
            s0 = wmma_bf16(qf[kk], kf0, s0);
            s1 = wmma_bf16(qf[kk], kf1, s1);
        }
#else
        const size_t kcol0 = (size_t)b * S + jb + (lane & 15);
#pragma unroll
        for (int kk = 0; kk < 4; ++kk) {
            bf16x16 kf0 = load_b_bf16(Kh, (size_t)Dm, kcol0,      32 * kk, lane);
            bf16x16 kf1 = load_b_bf16(Kh, (size_t)Dm, kcol0 + 16, 32 * kk, lane);
            s0 = wmma_bf16(qf[kk], kf0, s0);
            s1 = wmma_bf16(qf[kk], kf1, s1);
        }
#endif
#pragma unroll
        for (int e = 0; e < 8; ++e) { s0[e] *= sc; s1[e] *= sc; }

        // ---- online softmax (row = r + 8*(lane>>4); reduce over 16 lanes) ----
        float p0a[8], p1a[8], alpha[8];
#pragma unroll
        for (int r = 0; r < 8; ++r) {
            float t = fmaxf(s0[r], s1[r]);
            t = fmaxf(t, __shfl_xor(t, 1, 32));
            t = fmaxf(t, __shfl_xor(t, 2, 32));
            t = fmaxf(t, __shfl_xor(t, 4, 32));
            t = fmaxf(t, __shfl_xor(t, 8, 32));
            const float mnew = fmaxf(mrow[r], t);
            const float a  = __expf(mrow[r] - mnew);
            const float e0 = __expf(s0[r] - mnew);
            const float e1 = __expf(s1[r] - mnew);
            float su = e0 + e1;
            su += __shfl_xor(su, 1, 32);
            su += __shfl_xor(su, 2, 32);
            su += __shfl_xor(su, 4, 32);
            su += __shfl_xor(su, 8, 32);
            lrow[r] = lrow[r] * a + su;
            mrow[r] = mnew;
            alpha[r] = a;
            p0a[r] = e0; p1a[r] = e1;
        }
#pragma unroll
        for (int t = 0; t < 8; ++t)
#pragma unroll
            for (int r = 0; r < 8; ++r) accO[t][r] *= alpha[r];

        // ---- relayout P: C-layout regs -> LDS -> A-layout frag ----
        {
            const int rb = 8 * (lane >> 4), c = lane & 15;
#pragma unroll
            for (int r = 0; r < 8; ++r) {
                plds[wid][rb + r][c]      = (__bf16)p0a[r];
                plds[wid][rb + r][16 + c] = (__bf16)p1a[r];
            }
        }
        __syncthreads();
        bf16x16 pf = load_a_lds(&plds[wid][lane & 15][0], lane);

        // ---- O += P V  (8 d-tiles of 16 columns) ----
#if USE_TDM
#pragma unroll
        for (int t = 0; t < 8; ++t) {
            bf16x16 vf = load_b_strided(&vtile[cur][0][t * 16 + (lane & 15)], 136, lane);
            accO[t] = wmma_bf16(pf, vf, accO[t]);
        }
#else
        const __bf16* Vblk = Vm + ((size_t)b * S + jb) * Dm + h * hd;
#pragma unroll
        for (int t = 0; t < 8; ++t) {
            bf16x16 vf = load_b_strided(Vblk + t * 16 + (lane & 15), (size_t)Dm, lane);
            accO[t] = wmma_bf16(pf, vf, accO[t]);
        }
#endif
        __syncthreads();
    }

    const int mOff = 8 * (lane >> 4), nOff = (lane & 15);
#pragma unroll
    for (int t = 0; t < 8; ++t)
#pragma unroll
        for (int r = 0; r < 8; ++r) {
            const size_t row = (size_t)b * S + qb + r + mOff;
            const size_t col = (size_t)(h * hd + t * 16 + nOff);
            Om[row * Dm + col] = (__bf16)(accO[t][r] / lrow[r]);
        }
}

// ---------------------------------------------------------------------------
// Host launcher
// ---------------------------------------------------------------------------
extern "C" void kernel_launch(void* const* d_in, const int* in_sizes, int n_in,
                              void* d_out, int out_size, void* d_ws, size_t ws_size,
                              hipStream_t stream) {
    (void)in_sizes; (void)n_in; (void)out_size; (void)ws_size;

    const float* x      = (const float*)d_in[0];
    const float* wq     = (const float*)d_in[1];
    const float* wk     = (const float*)d_in[2];
    const float* wv     = (const float*)d_in[3];
    const float* wo     = (const float*)d_in[4];
    const float* w_up   = (const float*)d_in[5];
    const float* w_down = (const float*)d_in[6];
    const float* ln1w   = (const float*)d_in[7];
    const float* ln1b   = (const float*)d_in[8];
    const float* ln2w   = (const float*)d_in[9];
    const float* ln2b   = (const float*)d_in[10];
    float* out = (float*)d_out;

    const int B = 2, S = 2048, Dm = 2048, H = 16, DFF = 8192;
    const int M = B * S;
    const size_t act = (size_t)M * Dm;

    char* ws = (char*)d_ws;
    __bf16* h1 = (__bf16*)(ws);
    __bf16* qb = h1 + act;
    __bf16* kb = qb + act;
    __bf16* vb = kb + act;
    __bf16* ob = vb + act;
    __bf16* h2 = ob + act;
    __bf16* up = h2 + act;               // M * DFF elems

    const dim3 wblk(32, 4);

    ln_kernel<<<M, 256, 0, stream>>>(x, ln1w, ln1b, h1, Dm);
    gemm_nt<0><<<dim3(Dm / 128, M / 128), wblk, 0, stream>>>(h1, wq, M, Dm, Dm, nullptr, nullptr, qb);
    gemm_nt<0><<<dim3(Dm / 128, M / 128), wblk, 0, stream>>>(h1, wk, M, Dm, Dm, nullptr, nullptr, kb);
    gemm_nt<0><<<dim3(Dm / 128, M / 128), wblk, 0, stream>>>(h1, wv, M, Dm, Dm, nullptr, nullptr, vb);
    attn_kernel<<<dim3(S / 64, H, B), wblk, 0, stream>>>(qb, kb, vb, ob, S, Dm);
    gemm_nt<2><<<dim3(Dm / 128, M / 128), wblk, 0, stream>>>(ob, wo, M, Dm, Dm, x, out, nullptr);
    ln_kernel<<<M, 256, 0, stream>>>(out, ln2w, ln2b, h2, Dm);
    gemm_nt<1><<<dim3(DFF / 128, M / 128), wblk, 0, stream>>>(h2, w_up, M, DFF, Dm, nullptr, nullptr, up);
    gemm_nt<3><<<dim3(Dm / 128, M / 128), wblk, 0, stream>>>(up, w_down, M, Dm, DFF, nullptr, out, nullptr);
}
